// PointerAttentionDecoder_40072044871799
// MI455X (gfx1250) — matrix operand order repaired and checked by reading, not compiled, not run on hardware
//
#include <hip/hip_runtime.h>
#include <hip/hip_bf16.h>

#define B_   32
#define L_   300
#define T_   12
#define H_   256
#define E_   128
#define V_   50000
#define OOV_ 30
#define NPR_ 6
#define UNK_ 1

typedef __attribute__((ext_vector_type(16))) __bf16 v16bf;
typedef __attribute__((ext_vector_type(8)))  float  v8f;
typedef __attribute__((ext_vector_type(4))) unsigned int u32x4;
typedef __attribute__((ext_vector_type(8))) int          i32x8;
typedef __attribute__((ext_vector_type(4))) int          i32x4;

// LDS row stride for the TDM-staged A tile: 768 DWORDs + 1 pad DWORD per 256
// stored DWORDs (D# pad_interval=256, pad_amount=1) -> 771. 771 % 64 = 3,
// coprime with the 64 LDS banks, so 16 row-readers hit 16 distinct banks.
#define A_LDS_STRIDE 771

// ---------------------------------------------------------------------------
// WMMA fragment loaders: f32 memory -> bf16 fragments (RNE convert at load).
// Layouts used here (wave32; A/B in contiguous k = 16*half + s, same for both, not the ISA table order):
//   A 16x32 bf16: lane r (0..15) row M=r, holds K = 0..15; lanes 16..31
//                 same rows, K offset +16.  VGPR v packs K pair (kb, kb+1).
//   B 32x16 bf16: lane holds col N=lane%16; lanes 0..15 K=0..15, lanes 16..31
//                 K=16..31; VGPR v packs rows 2v, 2v+1 of the lane's K range.
//   C/D 16x16 f32: lane%16 = col, VGPR v = row v + 8*(lane/16).
// ---------------------------------------------------------------------------
__device__ __forceinline__ v16bf load_A_frag(const float* A, int lda, int m0, int k0) {
    int lane = threadIdx.x & 31;
    int r    = lane & 15;
    int half = lane >> 4;
    const float* row = A + (size_t)(m0 + r) * lda + k0;
    v16bf a;
#pragma unroll
    for (int v = 0; v < 8; ++v) {
        int kb = 2 * v + 16 * half;
        float2 p = *reinterpret_cast<const float2*>(row + kb);
        a[2 * v]     = (__bf16)p.x;
        a[2 * v + 1] = (__bf16)p.y;
    }
    return a;
}

// A-fragment from the TDM-staged, padded LDS tile (16 rows, stride 771 DWORDs,
// plus one pad DWORD after each 256 payload DWORDs within a row).
__device__ __forceinline__ v16bf load_A_frag_lds(const float* As, int k0) {
    int lane = threadIdx.x & 31;
    int r    = lane & 15;
    int half = lane >> 4;
    int base = r * A_LDS_STRIDE;
    v16bf a;
#pragma unroll
    for (int v = 0; v < 8; ++v) {
        int kb  = 2 * v + 16 * half;
        int k   = k0 + kb;
        int off = base + k + (k >> 8);      // account for intra-row pad DWORDs
        a[2 * v]     = (__bf16)As[off];
        a[2 * v + 1] = (__bf16)As[off + 1];
    }
    return a;
}

__device__ __forceinline__ v16bf load_B_frag(const float* Bm, int ldb, int k0, int n0) {
    int lane = threadIdx.x & 31;
    int c    = lane & 15;
    int half = lane >> 4;
    const float* col = Bm + (size_t)(k0 + 16 * half) * ldb + (n0 + c);
    v16bf b;
#pragma unroll
    for (int v = 0; v < 8; ++v) {
        b[2 * v]     = (__bf16)col[(size_t)(2 * v) * ldb];
        b[2 * v + 1] = (__bf16)col[(size_t)(2 * v + 1) * ldb];
    }
    return b;
}

// ---------------------------------------------------------------------------
// Generic one-wave-per-16x16-tile GEMM:  C[M,N] (+)= A[M,K] @ B[K,N] (+bias)
// K must be a multiple of 32; grid = (N/16, M/16), block = 32 (one wave).
// ---------------------------------------------------------------------------
__global__ void gemm16_bf16(const float* __restrict__ A, int lda,
                            const float* __restrict__ Bm, int ldb,
                            float* __restrict__ C, int ldc,
                            const float* __restrict__ bias,
                            int K, int accumulate) {
    int n0    = blockIdx.x * 16;
    int m0    = blockIdx.y * 16;
    int lane  = threadIdx.x & 31;
    int nn    = n0 + (lane & 15);
    int mbase = m0 + 8 * (lane >> 4);

    v8f acc = {};
    if (accumulate) {
#pragma unroll
        for (int v = 0; v < 8; ++v) acc[v] = C[(size_t)(mbase + v) * ldc + nn];
    }
    for (int k0 = 0; k0 < K; k0 += 32) {
        v16bf a = load_A_frag(A, lda, m0, k0);
        v16bf b = load_B_frag(Bm, ldb, k0, n0);
        acc = __builtin_amdgcn_wmma_f32_16x16x32_bf16(false, a, false, b,
                                                      (short)0, acc, false, false);
    }
    float bv = bias ? bias[nn] : 0.f;
#pragma unroll
    for (int v = 0; v < 8; ++v) C[(size_t)(mbase + v) * ldc + nn] = acc[v] + bv;
}

// ---------------------------------------------------------------------------
// Fused logits + streaming log-sum-exp over V (online softmax in registers).
// grid = (16, 24 row-tiles), block = 128 (4 waves). Wave 0 stages the 16x768
// f32 hc row-block into LDS via the Tensor Data Mover (one TDM descriptor,
// padded to avoid LDS bank conflicts), then all 4 waves stream V-tiles from
// the L2-resident Vout_w, doing bf16 WMMA with f32 accumulation and a
// cross-lane online softmax (shfl_xor over the 16 column lanes per row).
// ---------------------------------------------------------------------------
__global__ void vout_lse_kernel(const float* __restrict__ hc,
                                const float* __restrict__ Vw,
                                const float* __restrict__ Vb,
                                float* __restrict__ pm, float* __restrict__ ps) {
    __shared__ float hcs[16 * A_LDS_STRIDE];   // ~48.2 KB staged A row-block
    int m0   = blockIdx.y * 16;
    int wv   = blockIdx.x * 4 + (threadIdx.x >> 5);   // 0..63 global wave id
    int lane = threadIdx.x & 31;

    if (threadIdx.x < 32) {
        // Build the Tensor DMA Descriptor (CDNA5 ISA).
        unsigned long long ga = (unsigned long long)(const void*)(hc + (size_t)m0 * (3 * H_));
        unsigned int lds_base = (unsigned int)(size_t)hcs;  // low 32 bits = LDS offset
        u32x4 g0;
        g0[0] = 1u;                                   // count=1 (valid user D#)
        g0[1] = lds_base;                             // lds_addr
        g0[2] = (unsigned int)ga;                     // global_addr[31:0]
        g0[3] = (unsigned int)(ga >> 32) | (2u << 30);// global_addr[56:32] | type=2
        i32x8 g1;
        g1[0] = (2 << 16) | (1 << 20) | (7 << 22);    // data_size=4B, pad_enable,
                                                      // pad_interval=256 DW, pad_amount=1 DW
        g1[1] = (int)((3 * H_) << 16);                // tensor_dim0[15:0] = 768
        g1[2] = (int)(16 << 16);                      // tensor_dim0 hi=0 | tensor_dim1=16
        g1[3] = (int)((3 * H_) << 16);                // tensor_dim1 hi=0 | tile_dim0=768
        g1[4] = 16;                                   // tile_dim1=16, tile_dim2=0
        g1[5] = 3 * H_;                               // tensor_dim0_stride = 768
        g1[6] = 0;
        g1[7] = 0;
        i32x4 gz4 = {0, 0, 0, 0};                     // 2D tensor: groups 2/3 unused
        i32x8 gz8 = {0, 0, 0, 0, 0, 0, 0, 0};
        __builtin_amdgcn_tensor_load_to_lds(g0, g1, gz4, gz4, gz8, 0);
        __builtin_amdgcn_s_wait_tensorcnt(0);
    }
    __syncthreads();

    float m8[8], s8[8];
#pragma unroll
    for (int v = 0; v < 8; ++v) { m8[v] = -3.0e38f; s8[v] = 0.f; }

    for (int tile = wv; tile < V_ / 16; tile += 64) {
        int n0 = tile * 16;
        // speculative prefetch of the next tile's first B rows into L2
        if (tile + 64 < V_ / 16)
            __builtin_prefetch(Vw + (size_t)lane * V_ + (n0 + 64 * 16), 0, 1);
        v8f acc = {};
#pragma unroll 4
        for (int k0 = 0; k0 < 3 * H_; k0 += 32) {
            v16bf a = load_A_frag_lds(hcs, k0);
            v16bf b = load_B_frag(Vw, V_, k0, n0);
            acc = __builtin_amdgcn_wmma_f32_16x16x32_bf16(false, a, false, b,
                                                          (short)0, acc, false, false);
        }
        float bv = Vb[n0 + (lane & 15)];
#pragma unroll
        for (int v = 0; v < 8; ++v) {
            float xv = acc[v] + bv;
            // reduce across the 16 lanes holding this row's 16 columns
            float tmax = xv;
            tmax = fmaxf(tmax, __shfl_xor(tmax, 1, 32));
            tmax = fmaxf(tmax, __shfl_xor(tmax, 2, 32));
            tmax = fmaxf(tmax, __shfl_xor(tmax, 4, 32));
            tmax = fmaxf(tmax, __shfl_xor(tmax, 8, 32));
            float mnew = fmaxf(m8[v], tmax);
            float e = __expf(xv - mnew);
            e += __shfl_xor(e, 1, 32);
            e += __shfl_xor(e, 2, 32);
            e += __shfl_xor(e, 4, 32);
            e += __shfl_xor(e, 8, 32);
            s8[v] = s8[v] * __expf(m8[v] - mnew) + e;
            m8[v] = mnew;
        }
    }
    if ((lane & 15) == 0) {
        int half = lane >> 4;
#pragma unroll
        for (int v = 0; v < 8; ++v) {
            int rg = m0 + 8 * half + v;
            pm[rg * 64 + wv] = m8[v];
            ps[rg * 64 + wv] = s8[v];
        }
    }
}

// ----------------------------- small kernels -------------------------------
__device__ __forceinline__ float sigmoidf_(float x) { return 1.f / (1.f + __expf(-x)); }

__global__ void embed_kernel(const float* __restrict__ emb, const int* __restrict__ dec_input,
                             float* __restrict__ x) {
    int idx = blockIdx.x * blockDim.x + threadIdx.x;
    if (idx >= B_ * T_ * E_) return;
    int bt = idx / E_, e = idx % E_;
    x[idx] = emb[(size_t)dec_input[bt] * E_ + e];
}

__global__ void lstm_elem_kernel(const float* __restrict__ gates,
                                 const float* __restrict__ cprev,
                                 float* __restrict__ cnext, float* __restrict__ hnext,
                                 float* __restrict__ hidden, int t) {
    int idx = blockIdx.x * blockDim.x + threadIdx.x;
    if (idx >= B_ * H_) return;
    int b = idx / H_, h = idx % H_;
    const float* g = gates + (size_t)b * 4 * H_;
    float gi = g[h], gf = g[H_ + h], gg = g[2 * H_ + h], go = g[3 * H_ + h];
    float c  = sigmoidf_(gf) * cprev[idx] + sigmoidf_(gi) * tanhf(gg);
    float hn = sigmoidf_(go) * tanhf(c);
    cnext[idx] = c;
    hnext[idx] = hn;
    hidden[(size_t)(b * T_ + t) * H_ + h] = hn;
}

// e = tanh(enc_proj + dec_proj) . v_w + v_b, masked, softmax over L -> attn
__global__ void attn_kernel(const float* __restrict__ enc_proj,
                            const float* __restrict__ dec_proj,
                            const float* __restrict__ vw, const float* __restrict__ vb,
                            const unsigned char* __restrict__ mask,
                            float* __restrict__ attn) {
    __shared__ float dec[2 * H_];
    __shared__ float vws[2 * H_];
    __shared__ float ev[L_];
    __shared__ float red[128];
    int bt = blockIdx.x, b = bt / T_, tid = threadIdx.x;
    for (int d = tid; d < 2 * H_; d += 128) {
        dec[d] = dec_proj[(size_t)bt * 2 * H_ + d];
        vws[d] = vw[d];
    }
    __syncthreads();
    for (int l = tid; l < L_; l += 128) {
        const float* ep = enc_proj + (size_t)(b * L_ + l) * 2 * H_;
        float acc = 0.f;
        for (int d = 0; d < 2 * H_; ++d) acc += tanhf(ep[d] + dec[d]) * vws[d];
        acc += vb[0];
        if (mask[b * L_ + l]) acc = -1e30f;
        ev[l] = acc;
    }
    __syncthreads();
    float m = -1e30f;
    for (int l = tid; l < L_; l += 128) m = fmaxf(m, ev[l]);
    red[tid] = m; __syncthreads();
    for (int s = 64; s > 0; s >>= 1) { if (tid < s) red[tid] = fmaxf(red[tid], red[tid + s]); __syncthreads(); }
    m = red[0]; __syncthreads();
    float sum = 0.f;
    for (int l = tid; l < L_; l += 128) sum += __expf(ev[l] - m);
    red[tid] = sum; __syncthreads();
    for (int s = 64; s > 0; s >>= 1) { if (tid < s) red[tid] += red[tid + s]; __syncthreads(); }
    float inv = 1.f / red[0];
    for (int l = tid; l < L_; l += 128) attn[(size_t)bt * L_ + l] = __expf(ev[l] - m) * inv;
}

__global__ void context_kernel(const float* __restrict__ attn,
                               const float* __restrict__ enc_states,
                               float* __restrict__ context) {
    __shared__ float a[L_];
    int bt = blockIdx.x, b = bt / T_, tid = threadIdx.x;
    for (int l = tid; l < L_; l += 256) a[l] = attn[(size_t)bt * L_ + l];
    __syncthreads();
    for (int e = tid; e < 2 * H_; e += 256) {
        const float* es = enc_states + (size_t)b * L_ * 2 * H_ + e;
        float acc = 0.f;
        for (int l = 0; l < L_; ++l) acc += a[l] * es[(size_t)l * 2 * H_];
        context[(size_t)bt * 2 * H_ + e] = acc;
    }
}

__global__ void concat_hc_kernel(const float* __restrict__ hidden,
                                 const float* __restrict__ context,
                                 float* __restrict__ hc) {
    int idx = blockIdx.x * blockDim.x + threadIdx.x;
    if (idx >= B_ * T_ * 3 * H_) return;
    int bt = idx / (3 * H_), d = idx % (3 * H_);
    hc[idx] = (d < H_) ? hidden[(size_t)bt * H_ + d]
                       : context[(size_t)bt * 2 * H_ + (d - H_)];
}

__global__ void pswitch_kernel(const float* __restrict__ context, const float* __restrict__ hidden,
                               const float* __restrict__ x, const float* __restrict__ enc_h0,
                               const float* __restrict__ whw, const float* __restrict__ whb,
                               const float* __restrict__ wsw, const float* __restrict__ wsb,
                               const float* __restrict__ wxw, const float* __restrict__ wxb,
                               const float* __restrict__ wcw, const float* __restrict__ wcb,
                               float* __restrict__ psw) {
    __shared__ float red[128];
    __shared__ float sw[3];
    int bt = blockIdx.x, b = bt / T_, tid = threadIdx.x;
    for (int j = 0; j < 3; ++j) {
        float acc = 0.f;
        for (int d = tid; d < 2 * H_; d += 128) acc += context[(size_t)bt * 2 * H_ + d] * whw[d * 3 + j];
        for (int d = tid; d < H_;     d += 128) acc += hidden[(size_t)bt * H_ + d] * wsw[d * 3 + j];
        for (int d = tid; d < E_;     d += 128) acc += x[(size_t)bt * E_ + d] * wxw[d * 3 + j];
        for (int d = tid; d < H_;     d += 128) acc += enc_h0[(size_t)b * H_ + d] * wcw[d * 3 + j];
        red[tid] = acc; __syncthreads();
        for (int s = 64; s > 0; s >>= 1) { if (tid < s) red[tid] += red[tid + s]; __syncthreads(); }
        if (tid == 0) sw[j] = red[0] + whb[j] + wsb[j] + wxb[j] + wcb[j];
        __syncthreads();
    }
    if (tid == 0) {
        float m = fmaxf(sw[0], fmaxf(sw[1], sw[2]));
        float e0 = __expf(sw[0] - m), e1 = __expf(sw[1] - m), e2 = __expf(sw[2] - m);
        float inv = 1.f / (e0 + e1 + e2);
        psw[bt * 4 + 0] = e0 * inv;
        psw[bt * 4 + 1] = e1 * inv;
        psw[bt * 4 + 2] = e2 * inv;
    }
}

__global__ void tgt_logit_kernel(const float* __restrict__ hc, const float* __restrict__ Vw,
                                 const float* __restrict__ Vb, const int* __restrict__ targets,
                                 float* __restrict__ tgt) {
    __shared__ float red[256];
    int bt = blockIdx.x, tid = threadIdx.x;
    int z = targets[bt];
    float acc = 0.f;
    if (z < V_)
        for (int d = tid; d < 3 * H_; d += 256) acc += hc[(size_t)bt * 3 * H_ + d] * Vw[(size_t)d * V_ + z];
    red[tid] = acc; __syncthreads();
    for (int s = 128; s > 0; s >>= 1) { if (tid < s) red[tid] += red[tid + s]; __syncthreads(); }
    if (tid == 0) tgt[bt] = red[0] + ((z < V_) ? Vb[z] : 0.f);
}

__global__ void copysum_kernel(const float* __restrict__ attn, const int* __restrict__ article_inds,
                               const int* __restrict__ targets, float* __restrict__ cps) {
    __shared__ float red[128];
    int bt = blockIdx.x, b = bt / T_, tid = threadIdx.x;
    int z = targets[bt];
    float acc = 0.f;
    for (int l = tid; l < L_; l += 128)
        if (article_inds[b * L_ + l] == z) acc += attn[(size_t)bt * L_ + l];
    red[tid] = acc; __syncthreads();
    for (int s = 64; s > 0; s >>= 1) { if (tid < s) red[tid] += red[tid + s]; __syncthreads(); }
    if (tid == 0) cps[bt] = red[0];
}

__global__ void nll_kernel(const float* __restrict__ pm, const float* __restrict__ ps,
                           const float* __restrict__ tgtl, const float* __restrict__ cps,
                           const float* __restrict__ psw, const int* __restrict__ targets,
                           const float* __restrict__ dec_mask, float* __restrict__ nll) {
    __shared__ float red[64];
    int bt = blockIdx.x, tid = threadIdx.x;
    float m = pm[bt * 64 + tid];
    red[tid] = m; __syncthreads();
    for (int s = 32; s > 0; s >>= 1) { if (tid < s) red[tid] = fmaxf(red[tid], red[tid + s]); __syncthreads(); }
    float M = red[0]; __syncthreads();
    float e = ps[bt * 64 + tid] * __expf(m - M);
    red[tid] = e; __syncthreads();
    for (int s = 32; s > 0; s >>= 1) { if (tid < s) red[tid] += red[tid + s]; __syncthreads(); }
    if (tid == 0) {
        float lse = M + __logf(red[0]);
        int z = targets[bt];
        float out = psw[bt * 4 + 2] * cps[bt];
        if (z < V_) {
            float psel = (z < V_ - NPR_) ? psw[bt * 4 + 0] : psw[bt * 4 + 1];
            out += psel * __expf(tgtl[bt] - lse);
        }
        if (z == UNK_) out = 1.0f;
        nll[bt] = -__logf(out) * dec_mask[bt];
    }
}

__global__ void loss_kernel(const float* __restrict__ nll, const float* __restrict__ dec_lens,
                            float* __restrict__ out) {
    int b = threadIdx.x;
    if (b < B_) {
        float s = 0.f;
        for (int t = 0; t < T_; ++t) s += nll[b * T_ + t];
        out[b] = s / dec_lens[b];
    }
}

// ---------------------------------------------------------------------------
extern "C" void kernel_launch(void* const* d_in, const int* in_sizes, int n_in,
                              void* d_out, int out_size, void* d_ws, size_t ws_size,
                              hipStream_t stream) {
    (void)in_sizes; (void)n_in; (void)out_size; (void)ws_size;
    const float*         enc_states   = (const float*)d_in[0];
    const float*         enc_h0       = (const float*)d_in[1];
    const float*         enc_c0       = (const float*)d_in[2];
    const unsigned char* enc_mask     = (const unsigned char*)d_in[3];
    const int*           article_inds = (const int*)d_in[4];
    const int*           dec_input    = (const int*)d_in[5];
    const int*           targets      = (const int*)d_in[6];
    const float*         dec_lens     = (const float*)d_in[7];
    const float*         dec_mask     = (const float*)d_in[8];
    const float*         emb          = (const float*)d_in[9];
    const float*         W_ih         = (const float*)d_in[10];
    const float*         W_hh         = (const float*)d_in[11];
    const float*         b_lstm       = (const float*)d_in[12];
    const float*         Wh_w         = (const float*)d_in[13];
    const float*         Wh_b         = (const float*)d_in[14];
    const float*         Ws_w         = (const float*)d_in[15];
    const float*         Ws_b         = (const float*)d_in[16];
    const float*         v_w          = (const float*)d_in[17];
    const float*         v_b          = (const float*)d_in[18];
    const float*         wh_w         = (const float*)d_in[19];
    const float*         wh_b         = (const float*)d_in[20];
    const float*         ws_w         = (const float*)d_in[21];
    const float*         ws_b         = (const float*)d_in[22];
    const float*         wx_w         = (const float*)d_in[23];
    const float*         wx_b         = (const float*)d_in[24];
    const float*         wc_w         = (const float*)d_in[25];
    const float*         wc_b         = (const float*)d_in[26];
    const float*         Vout_w       = (const float*)d_in[27];
    const float*         Vout_b       = (const float*)d_in[28];

    float* w = (float*)d_ws;
    float* x        = w; w += B_ * T_ * E_;
    float* gates    = w; w += B_ * 4 * H_;
    float* hbuf0    = w; w += B_ * H_;
    float* hbuf1    = w; w += B_ * H_;
    float* cbuf0    = w; w += B_ * H_;
    float* cbuf1    = w; w += B_ * H_;
    float* hidden   = w; w += B_ * T_ * H_;
    float* enc_proj = w; w += B_ * L_ * 2 * H_;
    float* dec_proj = w; w += B_ * T_ * 2 * H_;
    float* attnw    = w; w += B_ * T_ * L_;
    float* contextw = w; w += B_ * T_ * 2 * H_;
    float* hc       = w; w += B_ * T_ * 3 * H_;
    float* psw      = w; w += B_ * T_ * 4;
    float* pm       = w; w += B_ * T_ * 64;
    float* ps       = w; w += B_ * T_ * 64;
    float* tgtl     = w; w += B_ * T_;
    float* cps      = w; w += B_ * T_;
    float* nll      = w; w += B_ * T_;

    embed_kernel<<<(B_ * T_ * E_ + 255) / 256, 256, 0, stream>>>(emb, dec_input, x);

    (void)hipMemcpyAsync(hbuf0, enc_h0, B_ * H_ * sizeof(float), hipMemcpyDeviceToDevice, stream);
    (void)hipMemcpyAsync(cbuf0, enc_c0, B_ * H_ * sizeof(float), hipMemcpyDeviceToDevice, stream);

    float* hb[2] = {hbuf0, hbuf1};
    float* cb[2] = {cbuf0, cbuf1};
    dim3 glstm(4 * H_ / 16, B_ / 16);
    for (int t = 0; t < T_; ++t) {
        int cur = t & 1, nxt = cur ^ 1;
        gemm16_bf16<<<glstm, 32, 0, stream>>>(x + t * E_, T_ * E_, W_ih, 4 * H_,
                                              gates, 4 * H_, b_lstm, E_, 0);
        gemm16_bf16<<<glstm, 32, 0, stream>>>(hb[cur], H_, W_hh, 4 * H_,
                                              gates, 4 * H_, nullptr, H_, 1);
        lstm_elem_kernel<<<(B_ * H_ + 255) / 256, 256, 0, stream>>>(gates, cb[cur], cb[nxt],
                                                                    hb[nxt], hidden, t);
    }

    gemm16_bf16<<<dim3(2 * H_ / 16, B_ * L_ / 16), 32, 0, stream>>>(
        enc_states, 2 * H_, Wh_w, 2 * H_, enc_proj, 2 * H_, Wh_b, 2 * H_, 0);
    gemm16_bf16<<<dim3(2 * H_ / 16, B_ * T_ / 16), 32, 0, stream>>>(
        hidden, H_, Ws_w, 2 * H_, dec_proj, 2 * H_, Ws_b, H_, 0);

    attn_kernel<<<B_ * T_, 128, 0, stream>>>(enc_proj, dec_proj, v_w, v_b, enc_mask, attnw);
    context_kernel<<<B_ * T_, 256, 0, stream>>>(attnw, enc_states, contextw);
    concat_hc_kernel<<<(B_ * T_ * 3 * H_ + 255) / 256, 256, 0, stream>>>(hidden, contextw, hc);
    pswitch_kernel<<<B_ * T_, 128, 0, stream>>>(contextw, hidden, x, enc_h0,
                                                wh_w, wh_b, ws_w, ws_b, wx_w, wx_b, wc_w, wc_b, psw);

    vout_lse_kernel<<<dim3(16, B_ * T_ / 16), 128, 0, stream>>>(hc, Vout_w, Vout_b, pm, ps);
    tgt_logit_kernel<<<B_ * T_, 256, 0, stream>>>(hc, Vout_w, Vout_b, targets, tgtl);
    copysum_kernel<<<B_ * T_, 128, 0, stream>>>(attnw, article_inds, targets, cps);
    nll_kernel<<<B_ * T_, 64, 0, stream>>>(pm, ps, tgtl, cps, psw, targets, dec_mask, nll);
    loss_kernel<<<1, 32, 0, stream>>>(nll, dec_lens, (float*)d_out);
}
